// VQCodebook_45268955300481
// MI455X (gfx1250) — matrix operand order repaired and checked by reading, not compiled, not run on hardware
//
#include <hip/hip_runtime.h>

typedef _Float16 v16h  __attribute__((ext_vector_type(16)));
typedef _Float16 half8 __attribute__((ext_vector_type(8)));
typedef _Float16 half4 __attribute__((ext_vector_type(4)));
typedef float    v8f   __attribute__((ext_vector_type(8)));
typedef float    f4    __attribute__((ext_vector_type(4)));

#define NUM_CODES   512
#define CODE_DIM    64
#define CB_STRIDE   72                      // halves per LDS codebook row (pad vs 64 to soften bank conflicts)
#define ROWS_BLK    256                     // rows per 256-thread block (8 waves x 32 rows)

#define LDS_CB_BYTES   (NUM_CODES * CB_STRIDE * 2)        // 73728
#define LDS_NORM_OFF   (LDS_CB_BYTES)                     // 512 * f32
#define LDS_IDX_OFF    (LDS_NORM_OFF + NUM_CODES * 4)     // 256 * i32
#define LDS_LOSS_OFF   (LDS_IDX_OFF + ROWS_BLK * 4)       // 1 * f32
#define LDS_TOTAL      (LDS_LOSS_OFF + 4)                 // 76804 bytes (< 320KB WGP LDS)

__global__ void vq_zero_kernel(float* lossOut) { *lossOut = 0.0f; }

__global__ __launch_bounds__(256) void vq_main_kernel(
    const float* __restrict__ zE, const float* __restrict__ cb,
    float* __restrict__ zq, float* __restrict__ idxOut,
    float* __restrict__ lossOut, float invCount) {
  extern __shared__ char smem[];
  _Float16* s_cb   = (_Float16*)smem;
  float*    s_nrm  = (float*)(smem + LDS_NORM_OFF);
  int*      s_idx  = (int*)(smem + LDS_IDX_OFF);
  float*    s_loss = (float*)(smem + LDS_LOSS_OFF);

  const int tid  = threadIdx.x;
  const int lane = tid & 31;
  const int wid  = tid >> 5;
  const int nl   = lane & 15;   // column / row-in-tile selector
  const int hf   = lane >> 4;   // which 16-lane half

  if (tid == 0) *s_loss = 0.0f;

  // ---- Phase 1: codebook (global f32) -> LDS f16 (padded rows) + f32 norms ----
  #pragma unroll
  for (int rr = 0; rr < 2; ++rr) {
    const int row = tid * 2 + rr;                    // 0..511
    const float* cr = cb + row * CODE_DIM;
    float nrm = 0.0f;
    #pragma unroll
    for (int k = 0; k < CODE_DIM; k += 4) {
      f4 v = *(const f4*)(cr + k);
      nrm += v.x * v.x + v.y * v.y + v.z * v.z + v.w * v.w;
      half4 h;
      h.x = (_Float16)v.x; h.y = (_Float16)v.y;
      h.z = (_Float16)v.z; h.w = (_Float16)v.w;
      *(half4*)(s_cb + row * CB_STRIDE + k) = h;     // ds_store_b64, 8B aligned
    }
    s_nrm[row] = nrm;
  }
  __syncthreads();

  // ---- Phase 2: load A fragments: 2 M-tiles x 2 K-fragments per wave ----
  // 16-bit A 16x32 layout: lanes 0-15 row M=nl hold K {kb..kb+7, kb+16..kb+23}, kb = hf*8
  const int rowBlock = blockIdx.x * ROWS_BLK;
  const int waveRow  = rowBlock + wid * 32;
  const int kb = hf * 8;

  v16h a[2][2];
  #pragma unroll
  for (int mt = 0; mt < 2; ++mt) {
    const float* zr = zE + (size_t)(waveRow + mt * 16 + nl) * CODE_DIM;
    #pragma unroll
    for (int kf = 0; kf < 2; ++kf) {
      const int o = kf * 32 + kb;
      f4 f0 = *(const f4*)(zr + o);
      f4 f1 = *(const f4*)(zr + o + 4);
      f4 f2 = *(const f4*)(zr + o + 16);
      f4 f3 = *(const f4*)(zr + o + 20);
      v16h t;
      t[0]=(_Float16)f0.x;  t[1]=(_Float16)f0.y;  t[2]=(_Float16)f0.z;  t[3]=(_Float16)f0.w;
      t[4]=(_Float16)f1.x;  t[5]=(_Float16)f1.y;  t[6]=(_Float16)f1.z;  t[7]=(_Float16)f1.w;
      t[8]=(_Float16)f2.x;  t[9]=(_Float16)f2.y;  t[10]=(_Float16)f2.z; t[11]=(_Float16)f2.w;
      t[12]=(_Float16)f3.x; t[13]=(_Float16)f3.y; t[14]=(_Float16)f3.z; t[15]=(_Float16)f3.w;
      a[mt][kf] = t;
    }
  }

  float bs[2][8];
  int   bi[2][8];
  #pragma unroll
  for (int mt = 0; mt < 2; ++mt)
    #pragma unroll
    for (int r = 0; r < 8; ++r) { bs[mt][r] = 3.4e38f; bi[mt][r] = 0; }

  // B 32x16 f16 layout: lane = N (0-15) / N (16-31), half hf holds K {kB..kB+7, kB+16..kB+23}, kB = hf*8
  const int kB = hf * 8;

  // ---- main loop over 32 N-tiles of 16 codes ----
  for (int n = 0; n < 32; ++n) {
    const int N = n * 16 + nl;
    const _Float16* bp = s_cb + N * CB_STRIDE;
    half8 b0lo = *(const half8*)(bp + kB);            // K kB..kB+7
    half8 b0hi = *(const half8*)(bp + kB + 16);       // K kB+16..kB+23
    half8 b1lo = *(const half8*)(bp + 32 + kB);       // K 32+kB..
    half8 b1hi = *(const half8*)(bp + 32 + kB + 16);
    v16h b0 = __builtin_shufflevector(b0lo, b0hi, 0,1,2,3,4,5,6,7,8,9,10,11,12,13,14,15);
    v16h b1 = __builtin_shufflevector(b1lo, b1hi, 0,1,2,3,4,5,6,7,8,9,10,11,12,13,14,15);
    const float nrm = s_nrm[N];

    #pragma unroll
    for (int mt = 0; mt < 2; ++mt) {
      v8f c = {0.f,0.f,0.f,0.f,0.f,0.f,0.f,0.f};
      c = __builtin_amdgcn_wmma_f32_16x16x32_f16(false, a[mt][0], false, b0,
                                                 (short)0, c, false, false);
      c = __builtin_amdgcn_wmma_f32_16x16x32_f16(false, a[mt][1], false, b1,
                                                 (short)0, c, false, false);
      // D layout: VGPR r -> row M = hf*8 + r, column N = nl-th of tile
      #pragma unroll
      for (int r = 0; r < 8; ++r) {
        const float sc = fmaf(-2.0f, c[r], nrm);      // ||e||^2 - 2 z.e
        const bool lt = sc < bs[mt][r];               // strict: first-occurrence ties
        bs[mt][r] = lt ? sc : bs[mt][r];
        bi[mt][r] = lt ? N  : bi[mt][r];
      }
    }
  }

  // ---- argmin reduction across the 16 lanes of each half (same M rows) ----
  #pragma unroll
  for (int m = 1; m < 16; m <<= 1) {
    #pragma unroll
    for (int mt = 0; mt < 2; ++mt) {
      #pragma unroll
      for (int r = 0; r < 8; ++r) {
        float os = __shfl_xor(bs[mt][r], m, 32);
        int   oi = __shfl_xor(bi[mt][r], m, 32);
        if (os < bs[mt][r] || (os == bs[mt][r] && oi < bi[mt][r])) {
          bs[mt][r] = os; bi[mt][r] = oi;
        }
      }
    }
  }

  if (nl == 0) {
    #pragma unroll
    for (int mt = 0; mt < 2; ++mt)
      #pragma unroll
      for (int r = 0; r < 8; ++r)
        s_idx[wid * 32 + mt * 16 + hf * 8 + r] = bi[mt][r];
  }
  __syncthreads();

  // ---- Phase 3: index output (float-cast), coalesced ----
  idxOut[rowBlock + tid] = (float)s_idx[tid];

  // ---- Phase 4: gather z_q (f32 codebook, L2-hot), write, exact f32 loss ----
  float lsum = 0.0f;
  #pragma unroll
  for (int it = 0; it < 16; ++it) {
    const int lr  = it * 16 + (tid >> 4);     // local row 0..255
    const int ci  = s_idx[lr];
    const int col = (tid & 15) * 4;
    f4 q = *(const f4*)(cb + ci * CODE_DIM + col);
    const size_t go = (size_t)(rowBlock + lr) * CODE_DIM + col;
    *(f4*)(zq + go) = q;
    f4 z = *(const f4*)(zE + go);
    f4 d = z - q;
    lsum += d.x * d.x + d.y * d.y + d.z * d.z + d.w * d.w;
  }
  #pragma unroll
  for (int m = 1; m < 32; m <<= 1) lsum += __shfl_xor(lsum, m, 32);
  if (lane == 0) atomicAdd(s_loss, lsum);
  __syncthreads();
  if (tid == 0) atomicAdd(lossOut, (*s_loss) * invCount);
}

extern "C" void kernel_launch(void* const* d_in, const int* in_sizes, int n_in,
                              void* d_out, int out_size, void* d_ws, size_t ws_size,
                              hipStream_t stream) {
  (void)n_in; (void)out_size; (void)d_ws; (void)ws_size;
  const float* zE = (const float*)d_in[0];
  const float* cb = (const float*)d_in[1];
  const int batch = in_sizes[0] / CODE_DIM;             // 1048576

  float* out     = (float*)d_out;
  float* zq      = out;                                 // batch * 64
  float* idxOut  = out + (size_t)batch * CODE_DIM;      // batch
  float* lossOut = idxOut + batch;                      // 1

  // Allow >64KB dynamic LDS (gfx1250 WGP has 320KB). Idempotent, non-stream op.
  (void)hipFuncSetAttribute((const void*)vq_main_kernel,
                            hipFuncAttributeMaxDynamicSharedMemorySize,
                            (int)LDS_TOTAL);

  vq_zero_kernel<<<1, 1, 0, stream>>>(lossOut);
  const int blocks = batch / ROWS_BLK;                  // 4096
  const float invCount = 1.0f / ((float)batch * (float)CODE_DIM);
  vq_main_kernel<<<blocks, 256, LDS_TOTAL, stream>>>(zE, cb, zq, idxOut, lossOut, invCount);
}
